// TaLKConvDecoderLayer_25391846654015
// MI455X (gfx1250) — compile-verified, matrix operand order checked by reading, not run on hardware
//
#include <hip/hip_runtime.h>

// ---------------- problem constants ----------------
#define Tn   4096
#define Bn   8
#define EMBc 1024
#define CONVc 1024
#define NH   16
#define FFNc 4096
#define KSc  31
#define ROWS (Tn * Bn)   // 32768 token rows

// ---------------- WMMA types ----------------
typedef __attribute__((ext_vector_type(16))) __bf16 v16bf;
typedef __attribute__((ext_vector_type(8)))  float  v8f;
typedef __attribute__((ext_vector_type(4)))  int    v4i;

// address-space-qualified int4 pointers for the async global->LDS builtin
typedef __attribute__((address_space(1))) v4i as1_v4i;
typedef __attribute__((address_space(3))) v4i as3_v4i;

// ---------------- CDNA5 async global->LDS (guarded) ----------------
#if defined(__has_builtin)
#if __has_builtin(__builtin_amdgcn_global_load_async_to_lds_b128) && \
    __has_builtin(__builtin_amdgcn_s_wait_asynccnt)
#define USE_ASYNC_LDS 1
#endif
#endif
#ifndef USE_ASYNC_LDS
#define USE_ASYNC_LDS 0
#endif

__device__ __forceinline__ unsigned short f2bf(float f) {
    unsigned int u = __builtin_bit_cast(unsigned int, f);
    unsigned int r = 0x7FFFu + ((u >> 16) & 1u);   // round-to-nearest-even
    return (unsigned short)((u + r) >> 16);
}
__device__ __forceinline__ float sigmoidf_(float x) {
    return 1.0f / (1.0f + __expf(-x));
}

// ---------------- fp32 -> bf16 convert (plain) ----------------
__global__ void f2bf_kernel(const float* __restrict__ src,
                            unsigned short* __restrict__ dst, int n) {
    int i = blockIdx.x * blockDim.x + threadIdx.x;
    if (i < n) dst[i] = f2bf(src[i]);
}

// ---------------- fp32 [K,N] -> bf16 [N,K] transpose-convert ----------------
// One-time on the (small) weight matrices so GEMM B staging becomes b128 ops.
__global__ void f2bfT_kernel(const float* __restrict__ src,
                             unsigned short* __restrict__ dst, int K, int N) {
    int i = blockIdx.x * blockDim.x + threadIdx.x;
    if (i < K * N) {
        int k = i / N;
        int n = i - k * N;
        dst[(size_t)n * K + k] = f2bf(src[i]);
    }
}

// ---------------- LayerNorm (fp32 in, bf16 out) ----------------
__global__ void __launch_bounds__(256)
ln_bf16_kernel(const float* __restrict__ x, const float* __restrict__ g,
               const float* __restrict__ b, unsigned short* __restrict__ out, int C) {
    __shared__ float red[256];
    const int row = blockIdx.x;
    const int tid = threadIdx.x;
    const float* xr = x + (size_t)row * C;

    float s = 0.f;
    for (int c = tid; c < C; c += 256) s += xr[c];
    red[tid] = s; __syncthreads();
    for (int st = 128; st > 0; st >>= 1) { if (tid < st) red[tid] += red[tid + st]; __syncthreads(); }
    const float mean = red[0] / (float)C;
    __syncthreads();

    float vs = 0.f;
    for (int c = tid; c < C; c += 256) { float d = xr[c] - mean; vs += d * d; }
    red[tid] = vs; __syncthreads();
    for (int st = 128; st > 0; st >>= 1) { if (tid < st) red[tid] += red[tid + st]; __syncthreads(); }
    const float rstd = rsqrtf(red[0] / (float)C + 1e-5f);

    unsigned short* orow = out + (size_t)row * C;
    for (int c = tid; c < C; c += 256)
        orow[c] = f2bf((xr[c] - mean) * rstd * g[c] + b[c]);
}

// ---------------- WMMA bf16 GEMM: 128x128 block tile, BK=64, 2-stage LDS ----------------
#define BK   64
#define LDSS 72   // row stride in ushort (144B: 16B-aligned, bank-conflict pad)

struct Frag2 { uint4 lo, hi; };

// Load one 16x32 bf16 operand tile from LDS ([16 rows][LDSS], pre-offset by k-slice)
// per the ISA 16-bit A/B layout: lanes 0-15 = rows 0-15 holding K{0..7,16..23};
// lanes 16-31 = rows 0-15 holding K{8..15,24..31}. Two b128 DS reads.
__device__ __forceinline__ v16bf load_frag(const unsigned short* tile, int lane) {
    const int row  = lane & 15;
    const int koff = (lane & 16) ? 8 : 0;
    const unsigned short* p = tile + row * LDSS + koff;
    Frag2 u;
    u.lo = *(const uint4*)(p);
    u.hi = *(const uint4*)(p + 16);
    return __builtin_bit_cast(v16bf, u);
}

// Stage a 128x64 bf16 tile (row-major, leading dim ld) into LDS.
// 256 threads x 4 16B chunks. Async path uses global_load_async_to_lds_b128.
__device__ __forceinline__ void stage_tile(const unsigned short* __restrict__ G,
                                           int ld, int row0, int k0,
                                           unsigned short* L, int tid) {
#pragma unroll
    for (int i = 0; i < 4; i++) {
        int idx = tid + i * 256;          // 0..1023
        int row = idx >> 3;               // 0..127
        int seg = (idx & 7) * 8;          // k offset 0,8,...,56
        const unsigned short* g = G + (size_t)(row0 + row) * ld + k0 + seg;
        unsigned short* l = L + row * LDSS + seg;
#if USE_ASYNC_LDS
        __builtin_amdgcn_global_load_async_to_lds_b128(
            (as1_v4i*)(unsigned long long)(uintptr_t)g,
            (as3_v4i*)(unsigned int)(uintptr_t)l,
            0, 0);
#else
        __builtin_prefetch(g + ld * BK, 0, 1);   // hint same k, BK rows further down
        *(uint4*)l = *(const uint4*)g;
#endif
    }
}

enum { EPI_STORE_F32 = 0, EPI_RESID_F32 = 1, EPI_SWISH_BF16 = 2 };

template <int EPI>
__global__ void __launch_bounds__(256)
gemm_bf16_kernel(const unsigned short* __restrict__ A,    // [M,K] bf16 row-major
                 const unsigned short* __restrict__ Bt,   // [N,K] bf16 row-major (pre-transposed)
                 const float* __restrict__ bias,          // [N]
                 const float* __restrict__ resid,         // [M,N] or null
                 float* __restrict__ Cf,
                 unsigned short* __restrict__ Cbf,
                 int M, int N, int K) {
    __shared__ unsigned short lds_a[2][128 * LDSS];
    __shared__ unsigned short lds_b[2][128 * LDSS];

    const int tid  = threadIdx.x;
    const int lane = tid & 31;
    const int wave = tid >> 5;
    const int wm   = (wave >> 2) * 64;   // wave M origin (2 wave-rows)
    const int wn   = (wave & 3) * 32;    // wave N origin (4 wave-cols)
    const int m0   = blockIdx.y * 128;
    const int n0   = blockIdx.x * 128;

    v8f zero = {};
    v8f acc[4][2];
#pragma unroll
    for (int mi = 0; mi < 4; mi++)
#pragma unroll
        for (int ni = 0; ni < 2; ni++) acc[mi][ni] = zero;

    const int nk = K / BK;

    // prologue: stage tile 0 into buffer 0
    stage_tile(A,  K, m0, 0, lds_a[0], tid);
    stage_tile(Bt, K, n0, 0, lds_b[0], tid);

    for (int kt = 0; kt < nk; kt++) {
        const int cur = kt & 1;
        if (kt + 1 < nk) {
            stage_tile(A,  K, m0, (kt + 1) * BK, lds_a[cur ^ 1], tid);
            stage_tile(Bt, K, n0, (kt + 1) * BK, lds_b[cur ^ 1], tid);
#if USE_ASYNC_LDS
            __builtin_amdgcn_s_wait_asynccnt(8);   // tile kt's 8 async ops done
#endif
        } else {
#if USE_ASYNC_LDS
            __builtin_amdgcn_s_wait_asynccnt(0);
#endif
        }
        __syncthreads();

        const unsigned short* La = lds_a[cur];
        const unsigned short* Lb = lds_b[cur];
#pragma unroll
        for (int kk = 0; kk < 2; kk++) {           // two 16x32 k-slices per stage
            v16bf af[4], bf[2];
#pragma unroll
            for (int mi = 0; mi < 4; mi++)
                af[mi] = load_frag(La + (wm + mi * 16) * LDSS + kk * 32, lane);
#pragma unroll
            for (int ni = 0; ni < 2; ni++)
                bf[ni] = load_frag(Lb + (wn + ni * 16) * LDSS + kk * 32, lane);
#pragma unroll
            for (int mi = 0; mi < 4; mi++)
#pragma unroll
                for (int ni = 0; ni < 2; ni++)
                    acc[mi][ni] = __builtin_amdgcn_wmma_f32_16x16x32_bf16(
                        false, af[mi], false, bf[ni], (short)0, acc[mi][ni], false, false);
        }
        __syncthreads();   // all waves done with buf[cur] before it is re-staged
    }

    // Epilogue: C/D layout — lanes 0-15: (M=r, N=lane); lanes 16-31: (M=r+8, N=lane-16)
    const int colLane = lane & 15;
    const int rowOff  = (lane & 16) ? 8 : 0;
#pragma unroll
    for (int mi = 0; mi < 4; mi++) {
#pragma unroll
        for (int ni = 0; ni < 2; ni++) {
            const int col  = n0 + wn + ni * 16 + colLane;
            const int rowb = m0 + wm + mi * 16 + rowOff;
            const float bv = bias[col];
#pragma unroll
            for (int r = 0; r < 8; r++) {
                const int row = rowb + r;
                float v = acc[mi][ni][r] + bv;
                const size_t o = (size_t)row * N + col;
                if (EPI == EPI_STORE_F32)      Cf[o] = v;
                else if (EPI == EPI_RESID_F32) Cf[o] = v + resid[o];
                else { v = v * sigmoidf_(v); Cbf[o] = f2bf(v); }   // swish -> bf16
            }
        }
    }
}

// ---------------- GLU: h = a * sigmoid(gate) ----------------
__global__ void __launch_bounds__(256)
glu_kernel(const float* __restrict__ h2, float* __restrict__ h) {
    size_t i  = (size_t)blockIdx.x * 256 + threadIdx.x;   // row*1024 + c
    size_t row = i >> 10;
    int c = (int)(i & 1023);
    float a  = h2[row * 2048 + c];
    float gt = h2[row * 2048 + 1024 + c];
    h[i] = a * sigmoidf_(gt);
}

// ---------------- alpha = sigmoid(h @ W_off + b_off), [T*B, 16] ----------------
__global__ void __launch_bounds__(256)
alpha_kernel(const float* __restrict__ h, const float* __restrict__ Woff,
             const float* __restrict__ boff, float* __restrict__ alpha) {
    __shared__ float red[256];
    const int row = blockIdx.x;          // t*B + b
    const int tid = threadIdx.x;
    const int g   = tid >> 4;            // head 0..15
    const int j   = tid & 15;
    const float* hr = h + (size_t)row * CONVc;
    float s = 0.f;
    for (int c = j; c < CONVc; c += 16) s += hr[c] * Woff[c * NH + g];
    red[tid] = s; __syncthreads();
    if (j == 0) {
        float t = 0.f;
#pragma unroll
        for (int i = 0; i < 16; i++) t += red[g * 16 + i];
        alpha[(size_t)row * NH + g] = sigmoidf_(t + boff[g]);   // == alpha[t*128 + b*16 + g]
    }
}

// ---------------- in-place causal cumsum over T (scale 1/(KSc+1)) ----------------
__global__ void __launch_bounds__(256)
cumsum_kernel(float* __restrict__ h) {
    const int col = blockIdx.x * 256 + threadIdx.x;   // 0..8191, coalesced per t-step
    const float scale = 1.0f / (float)(KSc + 1);
    float acc = 0.f;
    float* p = h + col;
    for (int t = 0; t < Tn; t++) {
        acc += p[(size_t)t * (Bn * CONVc)] * scale;
        p[(size_t)t * (Bn * CONVc)] = acc;
    }
}

// ---------------- TaLK gather: out = S[t] - lerp(S, t-1-alpha*eff) ----------------
__global__ void __launch_bounds__(256)
talk_gather_kernel(const float* __restrict__ S, const float* __restrict__ alpha,
                   unsigned short* __restrict__ out) {
    size_t idx = (size_t)blockIdx.x * 256 + threadIdx.x;  // t*8192 + b*1024 + c
    int c  = (int)(idx & (CONVc - 1));
    int b  = (int)((idx >> 10) & (Bn - 1));
    int t  = (int)(idx >> 13);
    int hh = c >> 6;                                       // R = 64
    float a   = alpha[(size_t)t * (Bn * NH) + b * NH + hh];
    float eff = (float)((t < KSc) ? t : KSc);
    float pos = (float)t - 1.0f - a * eff;
    float i0f = floorf(pos);
    float frac = pos - i0f;
    int i0 = (int)i0f;

    const float* Sc = S + (size_t)b * CONVc + c;
    const size_t stride = (size_t)Bn * CONVc;
    int c0 = (i0 < Tn - 1) ? i0 : (Tn - 1);
    int i1 = i0 + 1;
    int c1 = (i1 < Tn - 1) ? i1 : (Tn - 1);
    float v0 = (i0 >= 0) ? Sc[(size_t)c0 * stride] : 0.f;
    float v1 = (i1 >= 0) ? Sc[(size_t)c1 * stride] : 0.f;
    float st = Sc[(size_t)t * stride];
    out[idx] = f2bf(st - (v0 + frac * (v1 - v0)));
}

// ---------------- host-side orchestration ----------------
extern "C" void kernel_launch(void* const* d_in, const int* in_sizes, int n_in,
                              void* d_out, int out_size, void* d_ws, size_t ws_size,
                              hipStream_t stream) {
    const float* x      = (const float*)d_in[0];
    const float* W1     = (const float*)d_in[4];
    const float* b1     = (const float*)d_in[5];
    const float* Woff   = (const float*)d_in[6];
    const float* boff   = (const float*)d_in[7];
    const float* W2     = (const float*)d_in[8];
    const float* b2     = (const float*)d_in[9];
    const float* Wf1    = (const float*)d_in[10];
    const float* bf1    = (const float*)d_in[11];
    const float* Wf2    = (const float*)d_in[12];
    const float* bf2    = (const float*)d_in[13];
    const float* gconv  = (const float*)d_in[14];
    const float* beconv = (const float*)d_in[15];
    const float* gfin   = (const float*)d_in[16];
    const float* befin  = (const float*)d_in[17];
    float* out = (float*)d_out;

    // workspace arena (256B aligned slices)
    char* ws = (char*)d_ws;
    size_t off = 0;
    auto arena = [&](size_t bytes) -> char* {
        char* p = ws + off;
        off = (off + bytes + 255) & ~(size_t)255;
        return p;
    };
    unsigned short* W1b  = (unsigned short*)arena((size_t)EMBc * 2 * CONVc * 2);  // [2048,1024] N,K
    unsigned short* W2b  = (unsigned short*)arena((size_t)CONVc * EMBc * 2);      // [1024,1024]
    unsigned short* Wf1b = (unsigned short*)arena((size_t)EMBc * FFNc * 2);       // [4096,1024]
    unsigned short* Wf2b = (unsigned short*)arena((size_t)FFNc * EMBc * 2);       // [1024,4096]
    unsigned short* actb = (unsigned short*)arena((size_t)ROWS * CONVc * 2);      // xn -> conv -> yn
    float* big           = (float*)arena((size_t)ROWS * 2 * CONVc * 4);           // h2, reused as hf bf16
    float* h             = (float*)arena((size_t)ROWS * CONVc * 4);               // GLU out, then S in place
    float* y             = (float*)arena((size_t)ROWS * EMBc * 4);
    float* alpha         = (float*)arena((size_t)Tn * Bn * NH * 4);
    unsigned short* hfb  = (unsigned short*)big;

    // weights -> bf16, transposed to [N,K] so GEMM staging is pure b128 traffic
    f2bfT_kernel<<<(EMBc * 2 * CONVc + 255) / 256, 256, 0, stream>>>(W1, W1b, EMBc, 2 * CONVc);
    f2bfT_kernel<<<(CONVc * EMBc + 255) / 256, 256, 0, stream>>>(W2, W2b, CONVc, EMBc);
    f2bfT_kernel<<<(EMBc * FFNc + 255) / 256, 256, 0, stream>>>(Wf1, Wf1b, EMBc, FFNc);
    f2bfT_kernel<<<(FFNc * EMBc + 255) / 256, 256, 0, stream>>>(Wf2, Wf2b, FFNc, EMBc);

    // --- TaLK conv block ---
    ln_bf16_kernel<<<ROWS, 256, 0, stream>>>(x, gconv, beconv, actb, EMBc);

    dim3 g1(2 * CONVc / 128, ROWS / 128);
    gemm_bf16_kernel<EPI_STORE_F32><<<g1, 256, 0, stream>>>(
        actb, W1b, b1, nullptr, big, nullptr, ROWS, 2 * CONVc, EMBc);

    glu_kernel<<<(ROWS * CONVc) / 256, 256, 0, stream>>>(big, h);
    alpha_kernel<<<ROWS, 256, 0, stream>>>(h, Woff, boff, alpha);
    cumsum_kernel<<<(Bn * CONVc) / 256, 256, 0, stream>>>(h);
    talk_gather_kernel<<<(ROWS * CONVc) / 256, 256, 0, stream>>>(h, alpha, actb);

    dim3 g2(EMBc / 128, ROWS / 128);
    gemm_bf16_kernel<EPI_RESID_F32><<<g2, 256, 0, stream>>>(
        actb, W2b, b2, x, y, nullptr, ROWS, EMBc, CONVc);

    // --- FFN block ---
    ln_bf16_kernel<<<ROWS, 256, 0, stream>>>(y, gfin, befin, actb, EMBc);

    dim3 g3(FFNc / 128, ROWS / 128);
    gemm_bf16_kernel<EPI_SWISH_BF16><<<g3, 256, 0, stream>>>(
        actb, Wf1b, bf1, nullptr, nullptr, hfb, ROWS, FFNc, EMBc);

    dim3 g4(EMBc / 128, ROWS / 128);
    gemm_bf16_kernel<EPI_RESID_F32><<<g4, 256, 0, stream>>>(
        hfb, Wf2b, bf2, y, out, nullptr, ROWS, EMBc, FFNc);
}
